// ConfigurableSwitchedResidualGenerator3_37838661877906
// MI455X (gfx1250) — matrix operand order repaired and checked by reading, not compiled, not run on hardware
//
#include <hip/hip_runtime.h>

// ---------------------------------------------------------------------------
// Types
// ---------------------------------------------------------------------------
typedef __attribute__((ext_vector_type(16))) __bf16 v16bf;
typedef __attribute__((ext_vector_type(8)))  float  v8f;

static __device__ __forceinline__ float bf2f(__bf16 x) { return (float)x; }

// CDNA5 async memory->LDS copy (CDNA5 ISA): each lane moves 16B,
// tracked with ASYNCcnt; no VGPR data round-trip.
static __device__ __forceinline__ void async_load16(unsigned lds_off, const void* gptr) {
  asm volatile("global_load_async_to_lds_b128 %0, %1, off"
               :: "v"(lds_off), "v"(gptr)
               : "memory");
}
static __device__ __forceinline__ void wait_async0() {
  asm volatile("s_wait_asynccnt 0x0" ::: "memory");
}

// ---------------------------------------------------------------------------
// Weight packing: OIHW f32 -> WMMA A-fragment bf16 layout.
// Fragment index ((tap*KB + kb)*MB + mb), 512 bf16 per fragment
// (32 lanes x 16 elems).  Contiguous-k order used here (16-bit A 16x32; B uses the same):
//   m = mb*16 + (lane & 15)
//   k = kb*32 + (lane>>4)*16 + j
// Out-of-range (m>=Cout || k>=Cin) -> 0, which makes channel padding inert.
// ---------------------------------------------------------------------------
__global__ void pack_w_kernel(const float* __restrict__ w, __bf16* __restrict__ dst,
                              int Cout, int Cin, int CinP, int CoutP) {
  int idx = blockIdx.x * blockDim.x + threadIdx.x;
  int total = 9 * CinP * CoutP;
  if (idx >= total) return;
  int j    = idx & 15;
  int lane = (idx >> 4) & 31;
  int frag = idx >> 9;
  int MB = CoutP >> 4, KB = CinP >> 5;
  int mb  = frag % MB;
  int rest = frag / MB;
  int kb  = rest % KB;
  int tap = rest / KB;
  int m = mb * 16 + (lane & 15);
  int k = kb * 32 + ((lane >> 4) << 4) + j;
  float v = 0.f;
  if (m < Cout && k < Cin) v = w[((size_t)(m * Cin + k)) * 9 + tap];
  dst[idx] = (__bf16)v;
}

// ---------------------------------------------------------------------------
// NCHW f32 -> HWC bf16 with channel padding
// ---------------------------------------------------------------------------
__global__ void nchw2hwc_kernel(const float* __restrict__ src, __bf16* __restrict__ dst,
                                int B, int C, int CPad, int HW) {
  int idx = blockIdx.x * blockDim.x + threadIdx.x;
  int total = B * HW * CPad;
  if (idx >= total) return;
  int c = idx % CPad;
  int p = (idx / CPad) % HW;
  int b = idx / (CPad * HW);
  float v = (c < C) ? src[((size_t)b * C + c) * HW + p] : 0.f;
  dst[idx] = (__bf16)v;
}

// ---------------------------------------------------------------------------
// Implicit-GEMM 3x3 SAME conv, bf16 WMMA, f32 accumulate.
// Block tile: 32 Cout x (8 rows x 16 cols) pixels. 8 waves:
//   wave: mfrag = wv&1 (16 Cout rows), rows f0=(wv>>1)*2, f0+1.
// LDS: input halo tile 10 x 18 x CT (CT<=128) in HWC bf16.
// Interior tiles stage the halo via ASYNC global->LDS copies (ASYNCcnt);
// border tiles use the VGPR path with zero-fill for SAME padding.
// act: 0 = none, 1 = lrelu(0.2), 2 = silu
// out (bf16 HWC, CoutP channels) and/or out_nchw (f32, Cout channels).
// ---------------------------------------------------------------------------
#define TROWS 8
#define TCOLS 16
#define IR (TROWS + 2)
#define IC (TCOLS + 2)
#define CMAX 128

__global__ __launch_bounds__(256)
void conv3x3_wmma_kernel(const __bf16* __restrict__ in, const __bf16* __restrict__ wf,
                         const float* __restrict__ bias,
                         __bf16* __restrict__ out, float* __restrict__ out_nchw,
                         int H, int W, int CinP, int CoutP, int Cout, int act) {
  __shared__ __align__(16) __bf16 smem[IR * IC * CMAX];

  const int b = blockIdx.z;
  const int tilesPerRow = W / TCOLS;
  const int row0 = (blockIdx.x / tilesPerRow) * TROWS;
  const int col0 = (blockIdx.x % tilesPerRow) * TCOLS;
  const int tid  = threadIdx.x;
  const int lane = tid & 31, wv = tid >> 5;
  const int mfrag = wv & 1;
  const int f0 = (wv >> 1) * 2;
  const int MB = CoutP >> 4;
  const int KBtot = CinP >> 5;
  const int mbIdx = blockIdx.y * 2 + mfrag;
  const bool interior = (row0 > 0) && (row0 + TROWS < H) && (col0 > 0) && (col0 + TCOLS < W);
  const unsigned smem_base = (unsigned)(unsigned long long)(void*)smem;  // LDS offset = addr[31:0]

  v8f acc0 = {0.f, 0.f, 0.f, 0.f, 0.f, 0.f, 0.f, 0.f};
  v8f acc1 = {0.f, 0.f, 0.f, 0.f, 0.f, 0.f, 0.f, 0.f};

  for (int c0 = 0; c0 < CinP; c0 += CMAX) {
    const int CT = (CinP - c0) < CMAX ? (CinP - c0) : CMAX;  // multiple of 32
    const int vecsPerRow = (IC * CT) >> 3;
    const int vecs = IR * vecsPerRow;
    if (interior) {
      // ---- async global->LDS staging (no OOB possible) ------------------
      for (int e = tid; e < vecs; e += 256) {
        int r = e / vecsPerRow;
        int rem = e - r * vecsPerRow;
        int c = rem / (CT >> 3);
        int ch8 = rem - c * (CT >> 3);
        int gr = row0 + r - 1, gc = col0 + c - 1;
        const void* g = (const void*)(in + (((size_t)b * H + gr) * W + gc) * CinP + c0 + ch8 * 8);
        async_load16(smem_base + (unsigned)e * 16u, g);
      }
      wait_async0();
    } else {
      // ---- VGPR staging with zero-fill (SAME padding at borders) --------
      for (int e = tid; e < vecs; e += 256) {
        int r = e / vecsPerRow;
        int rem = e - r * vecsPerRow;
        int c = rem / (CT >> 3);
        int ch8 = rem - c * (CT >> 3);
        int gr = row0 + r - 1, gc = col0 + c - 1;
        uint4 v = make_uint4(0u, 0u, 0u, 0u);
        if ((unsigned)gr < (unsigned)H && (unsigned)gc < (unsigned)W) {
          const uint4* p = (const uint4*)(in + (((size_t)b * H + gr) * W + gc) * CinP + c0 + ch8 * 8);
          v = *p;
        }
        ((uint4*)smem)[e] = v;
      }
    }
    __syncthreads();

    // ---- GEMM over taps x K-blocks --------------------------------------
    const int n = lane & 15;
    const int half16 = (lane >> 4) * 16;
    const int kbs = CT >> 5;
    for (int tap = 0; tap < 9; ++tap) {
      const int dy = tap / 3, dx = tap - dy * 3;
      for (int kb = 0; kb < kbs; ++kb) {
        const int kbG = (c0 >> 5) + kb;
        v16bf a = *(const v16bf*)(wf + ((size_t)((tap * KBtot + kbG) * MB + mbIdx)) * 512 + lane * 16);
        const __bf16* brow0 = smem + ((f0 + dy) * IC + (n + dx)) * CT + kb * 32 + half16;
        const __bf16* brow1 = smem + ((f0 + 1 + dy) * IC + (n + dx)) * CT + kb * 32 + half16;
        v16bf b0 = *(const v16bf*)brow0;
        v16bf b1 = *(const v16bf*)brow1;
        acc0 = __builtin_amdgcn_wmma_f32_16x16x32_bf16(false, a, false, b0, (short)0, acc0, false, false);
        acc1 = __builtin_amdgcn_wmma_f32_16x16x32_bf16(false, a, false, b1, (short)0, acc1, false, false);
      }
    }
    __syncthreads();
  }

  // ---- epilogue: D layout lane=(N), vgpr v -> M = v + 8*(lane>=16) ------
  const int n = lane & 15;
  const int coutBase = blockIdx.y * 32 + mfrag * 16 + (lane >> 4) * 8;  // 8 consecutive ch
  for (int fi = 0; fi < 2; ++fi) {
    v8f acc = fi ? acc1 : acc0;
    int orow = row0 + f0 + fi;
    int ocol = col0 + n;
    alignas(16) unsigned short packed[8];
    for (int v = 0; v < 8; ++v) {
      float x = acc[v];
      int co = coutBase + v;
      if (bias) x += bias[co < Cout ? co : 0];
      if (act == 1) x = x >= 0.f ? x : 0.2f * x;
      else if (act == 2) x = x / (1.f + __expf(-x));
      if (out_nchw && co < Cout)
        out_nchw[(((size_t)b * Cout + co) * H + orow) * W + ocol] = x;
      __bf16 bv = (__bf16)x;
      packed[v] = __builtin_bit_cast(unsigned short, bv);
    }
    if (out) {
      __bf16* dst = out + (((size_t)b * H + orow) * W + ocol) * CoutP + coutBase;
      *(uint4*)dst = *(const uint4*)packed;
    }
  }
}

// ---------------------------------------------------------------------------
// GroupNorm (+optional SiLU), in-place on HWC bf16; one block per (b, group)
// ---------------------------------------------------------------------------
__global__ __launch_bounds__(256)
void gn_silu_kernel(__bf16* __restrict__ x, const float* __restrict__ g,
                    const float* __restrict__ be, int C, int CPad, int HW,
                    int groups, int dosilu) {
  __shared__ float sh[512];
  int b = blockIdx.x / groups;
  int grp = blockIdx.x % groups;
  int cs = C / groups;
  size_t base = (size_t)b * HW * CPad;
  int cnt = HW * cs;
  float s = 0.f, s2 = 0.f;
  for (int i = threadIdx.x; i < cnt; i += 256) {
    int p = i / cs;
    int c = grp * cs + (i - p * cs);
    float v = bf2f(x[base + (size_t)p * CPad + c]);
    s += v; s2 += v * v;
  }
  sh[threadIdx.x] = s; sh[256 + threadIdx.x] = s2;
  __syncthreads();
  for (int st = 128; st > 0; st >>= 1) {
    if (threadIdx.x < st) {
      sh[threadIdx.x] += sh[threadIdx.x + st];
      sh[256 + threadIdx.x] += sh[256 + threadIdx.x + st];
    }
    __syncthreads();
  }
  float mean = sh[0] / (float)cnt;
  float var = sh[256] / (float)cnt - mean * mean;
  float inv = rsqrtf(var + 1e-5f);
  __syncthreads();
  for (int i = threadIdx.x; i < cnt; i += 256) {
    int p = i / cs;
    int c = grp * cs + (i - p * cs);
    size_t o = base + (size_t)p * CPad + c;
    float v = (bf2f(x[o]) - mean) * inv * g[c] + be[c];
    if (dosilu) v = v / (1.f + __expf(-v));
    x[o] = (__bf16)v;
  }
}

// ---------------------------------------------------------------------------
// Nearest 2x upsample, HWC bf16
// ---------------------------------------------------------------------------
__global__ void up2_kernel(const __bf16* __restrict__ in, __bf16* __restrict__ out,
                           int B, int CPad, int H, int W) {
  int idx = blockIdx.x * blockDim.x + threadIdx.x;
  int H2 = H * 2, W2 = W * 2;
  size_t total = (size_t)B * H2 * W2 * CPad;
  if ((size_t)idx >= total) return;
  int c = idx % CPad;
  int xx = (idx / CPad) % W2;
  int yy = (idx / (CPad * W2)) % H2;
  int b = idx / (CPad * W2 * H2);
  out[idx] = in[(((size_t)b * H + (yy >> 1)) * W + (xx >> 1)) * CPad + c];
}

// ---------------------------------------------------------------------------
// xs = x0 + noise(NCHW f32) * noise_scale   (HWC bf16 out)
// ---------------------------------------------------------------------------
__global__ void add_noise_kernel(const __bf16* __restrict__ x0, const float* __restrict__ noise,
                                 const float* __restrict__ ns, __bf16* __restrict__ out,
                                 int B, int C, int CPad, int HW) {
  int idx = blockIdx.x * blockDim.x + threadIdx.x;
  int total = B * HW * CPad;
  if (idx >= total) return;
  int c = idx % CPad;
  int p = (idx / CPad) % HW;
  int b = idx / (CPad * HW);
  float v = 0.f;
  if (c < C) v = bf2f(x0[idx]) + noise[((size_t)b * C + c) * HW + p] * ns[0];
  out[idx] = (__bf16)v;
}

// ---------------------------------------------------------------------------
// att = softmax(m / 20) over T=8 channels (m HWC bf16 CPad=32) -> f32 [B,HW,8]
// ---------------------------------------------------------------------------
__global__ void softmax8_kernel(const __bf16* __restrict__ m, float* __restrict__ att,
                                int BHW, int CPad) {
  int idx = blockIdx.x * blockDim.x + threadIdx.x;
  if (idx >= BHW) return;
  const __bf16* p = m + (size_t)idx * CPad;
  float v[8], mx = -1e30f;
  for (int t = 0; t < 8; ++t) { v[t] = bf2f(p[t]) * (1.f / 20.f); mx = v[t] > mx ? v[t] : mx; }
  float sum = 0.f;
  for (int t = 0; t < 8; ++t) { v[t] = __expf(v[t] - mx); sum += v[t]; }
  float inv = 1.f / sum;
  for (int t = 0; t < 8; ++t) att[(size_t)idx * 8 + t] = v[t] * inv;
}

// ---------------------------------------------------------------------------
// acc (+)= (h4*tscale[t] + tbias[t]) * att[:,t]   acc f32 [B,HW,64]
// ---------------------------------------------------------------------------
__global__ void combine_kernel(const __bf16* __restrict__ h4, const float* __restrict__ att,
                               const float* __restrict__ tscale, const float* __restrict__ tbias,
                               int t, float* __restrict__ acc, int BHW, int C, int CPad, int init) {
  int idx = blockIdx.x * blockDim.x + threadIdx.x;
  int total = BHW * C;
  if (idx >= total) return;
  int c = idx % C;
  int p = idx / C;
  float v = bf2f(h4[(size_t)p * CPad + c]) * tscale[t] + tbias[t];
  float contrib = v * att[(size_t)p * 8 + t];
  if (init) acc[idx] = contrib; else acc[idx] += contrib;
}

// out0 = x0 + acc * switch_scale
__global__ void resid_switch_kernel(const __bf16* __restrict__ x0, const float* __restrict__ acc,
                                    const float* __restrict__ ss, __bf16* __restrict__ out,
                                    int BHW, int C, int CPad) {
  int idx = blockIdx.x * blockDim.x + threadIdx.x;
  int total = BHW * CPad;
  if (idx >= total) return;
  int c = idx % CPad;
  int p = idx / CPad;
  float v = 0.f;
  if (c < C) v = bf2f(x0[idx]) + acc[(size_t)p * C + c] * ss[0];
  out[idx] = (__bf16)v;
}

// out = base + t * s
__global__ void resid_scale_kernel(const __bf16* __restrict__ base, const __bf16* __restrict__ t,
                                   const float* __restrict__ s, __bf16* __restrict__ out, int total) {
  int idx = blockIdx.x * blockDim.x + threadIdx.x;
  if (idx >= total) return;
  out[idx] = (__bf16)(bf2f(base[idx]) + bf2f(t[idx]) * s[0]);
}

// ---------------------------------------------------------------------------
// Host orchestration
// ---------------------------------------------------------------------------
static inline int pad32(int c) { return (c + 31) & ~31; }

extern "C" void kernel_launch(void* const* d_in, const int* in_sizes, int n_in,
                              void* d_out, int out_size, void* d_ws, size_t ws_size,
                              hipStream_t stream) {
  const float* in_x      = (const float*)d_in[0];
  const float* in_spine  = (const float*)d_in[1];
  const float* in_noise  = (const float*)d_in[2];
  const float* init_w    = (const float*)d_in[3];
  const float* init_b    = (const float*)d_in[4];
  const float* pre_w     = (const float*)d_in[5];
  const float* tw0       = (const float*)d_in[6];
  const float* tw1       = (const float*)d_in[7];
  const float* tw2       = (const float*)d_in[8];
  const float* tw3       = (const float*)d_in[9];
  const float* tscale    = (const float*)d_in[10];
  const float* tbias     = (const float*)d_in[11];
  const float* noise_sc  = (const float*)d_in[12];
  const float* switch_sc = (const float*)d_in[13];
  const float* psc_sc    = (const float*)d_in[14];
  const float* post_w    = (const float*)d_in[15];
  const float* post_b    = (const float*)d_in[16];
  const float* m1_w      = (const float*)d_in[17];
  const float* m1_b      = (const float*)d_in[18];
  const float* m1_g      = (const float*)d_in[19];
  const float* m1_be     = (const float*)d_in[20];
  const float* m2_w      = (const float*)d_in[21];
  const float* m2_g      = (const float*)d_in[22];
  const float* m2_be     = (const float*)d_in[23];
  const float* u1a_w     = (const float*)d_in[24];
  const float* u1b_w     = (const float*)d_in[25];
  const float* u1b_g     = (const float*)d_in[26];
  const float* u1b_be    = (const float*)d_in[27];
  const float* u2a_w     = (const float*)d_in[28];
  const float* u2b_w     = (const float*)d_in[29];
  const float* u2b_g     = (const float*)d_in[30];
  const float* u2b_be    = (const float*)d_in[31];
  const float* mf_w      = (const float*)d_in[32];
  const float* up1_w     = (const float*)d_in[33];
  const float* up1_b     = (const float*)d_in[34];
  const float* up2_w     = (const float*)d_in[35];
  const float* up2_b     = (const float*)d_in[36];
  const float* hr_w      = (const float*)d_in[37];
  const float* hr_b      = (const float*)d_in[38];
  const float* fin_w     = (const float*)d_in[39];
  const float* fin_b     = (const float*)d_in[40];

  const int B = 16, F = 64, Fm = 96, T = 8;
  const int H0 = 64, W0 = 64, HW0 = H0 * W0;             // 4096
  const int H1 = 128, HW1 = H1 * H1;                     // 16384
  const int H2 = 256, HW2 = H2 * H2;                     // 65536
  const int Hs = 16, HWs = Hs * Hs;                      // spine 16x16
  const int Hm = 32, HWm = Hs * 2 * Hs * 2;              // 32x32

  char* ws = (char*)d_ws;
  size_t off = 0;
  auto alloc = [&](size_t bytes) -> char* {
    char* p = ws + off;
    off += (bytes + 255) & ~(size_t)255;
    return p;
  };
  auto wbytes = [](int Cout, int Cin) -> size_t {
    return (size_t)9 * pad32(Cin) * pad32(Cout) * sizeof(__bf16);
  };

  // ---- packed weight buffers ----
  char* w_init = alloc(wbytes(F, 3));
  char* w_pre  = alloc(wbytes(F, F));
  char* w_t0[8], *w_t1[8], *w_t2[8], *w_t3[8];
  for (int t = 0; t < T; ++t) {
    w_t0[t] = alloc(wbytes(Fm, F));
    w_t1[t] = alloc(wbytes(Fm, Fm));
    w_t2[t] = alloc(wbytes(Fm, Fm));
    w_t3[t] = alloc(wbytes(F, Fm));
  }
  char* w_post = alloc(wbytes(F, F));
  char* w_m1   = alloc(wbytes(256, 256));
  char* w_m2   = alloc(wbytes(256, 256));
  char* w_u1a  = alloc(wbytes(128, 256));
  char* w_u1b  = alloc(wbytes(128, 128));
  char* w_u2a  = alloc(wbytes(F, 128));
  char* w_u2b  = alloc(wbytes(F, F));
  char* w_mf   = alloc(wbytes(T, F));
  char* w_up1  = alloc(wbytes(F, F));
  char* w_up2  = alloc(wbytes(F, F));
  char* w_hr   = alloc(wbytes(F, F));
  char* w_fin  = alloc(wbytes(3, F));

  // ---- activation buffers (HWC bf16 unless noted) ----
  char* a_xin  = alloc((size_t)B * HW0 * 32 * 2);
  char* a_x0   = alloc((size_t)B * HW0 * 64 * 2);
  char* a_xs   = alloc((size_t)B * HW0 * 64 * 2);
  char* a_xs2  = alloc((size_t)B * HW0 * 64 * 2);
  char* a_h1   = alloc((size_t)B * HW0 * 96 * 2);
  char* a_h2   = alloc((size_t)B * HW0 * 96 * 2);
  char* a_h4   = alloc((size_t)B * HW0 * 64 * 2);
  char* a_acc  = alloc((size_t)B * HW0 * 64 * 4);  // f32
  char* a_sp   = alloc((size_t)B * HWs * 256 * 2);
  char* a_mA   = alloc((size_t)B * HWs * 256 * 2);
  char* a_mB   = alloc((size_t)B * HWs * 256 * 2);
  char* a_mu1  = alloc((size_t)B * HWm * 256 * 2);
  char* a_m3   = alloc((size_t)B * HWm * 128 * 2);
  char* a_m4   = alloc((size_t)B * HWm * 128 * 2);
  char* a_mu2  = alloc((size_t)B * HW0 * 128 * 2);
  char* a_m5   = alloc((size_t)B * HW0 * 64 * 2);
  char* a_m6   = alloc((size_t)B * HW0 * 64 * 2);
  char* a_mm   = alloc((size_t)B * HW0 * 32 * 2);
  char* a_att  = alloc((size_t)B * HW0 * 8 * 4);   // f32
  char* a_out0 = alloc((size_t)B * HW0 * 64 * 2);
  char* a_pT   = alloc((size_t)B * HW0 * 64 * 2);
  char* a_out1 = alloc((size_t)B * HW0 * 64 * 2);
  char* a_u1   = alloc((size_t)B * HW1 * 64 * 2);
  char* a_c1   = alloc((size_t)B * HW1 * 64 * 2);
  char* a_u2   = alloc((size_t)B * HW2 * 64 * 2);
  char* a_c2   = alloc((size_t)B * HW2 * 64 * 2);
  char* a_c3   = a_u2;  // reuse: u2 fully consumed before hr conv writes c3
  (void)ws_size; (void)in_sizes; (void)n_in; (void)out_size;

  // ---- helpers ----
  auto packW = [&](const float* w, char* dst, int Cout, int Cin) {
    int CinP = pad32(Cin), CoutP = pad32(Cout);
    int total = 9 * CinP * CoutP;
    pack_w_kernel<<<(total + 255) / 256, 256, 0, stream>>>(w, (__bf16*)dst, Cout, Cin, CinP, CoutP);
  };
  auto conv = [&](const char* in, const char* wf, const float* bias, char* outB,
                  float* outN, int H, int W, int CinP, int Cout, int act) {
    int CoutP = pad32(Cout);
    dim3 g((H / TROWS) * (W / TCOLS), CoutP / 32, B);
    conv3x3_wmma_kernel<<<g, 256, 0, stream>>>((const __bf16*)in, (const __bf16*)wf, bias,
                                               (__bf16*)outB, outN, H, W, CinP, CoutP, Cout, act);
  };
  auto gnSilu = [&](char* x, const float* g, const float* be, int C, int CPad, int HW) {
    gn_silu_kernel<<<B * 8, 256, 0, stream>>>((__bf16*)x, g, be, C, CPad, HW, 8, 1);
  };
  auto up2x = [&](const char* in, char* out, int CPad, int H, int W) {
    size_t total = (size_t)B * H * 2 * W * 2 * CPad;
    up2_kernel<<<(unsigned)((total + 255) / 256), 256, 0, stream>>>((const __bf16*)in, (__bf16*)out, B, CPad, H, W);
  };

  // ---- pack all weights ----
  packW(init_w, w_init, F, 3);
  packW(pre_w, w_pre, F, F);
  for (int t = 0; t < T; ++t) {
    packW(tw0 + (size_t)t * Fm * F * 9,  w_t0[t], Fm, F);
    packW(tw1 + (size_t)t * Fm * Fm * 9, w_t1[t], Fm, Fm);
    packW(tw2 + (size_t)t * Fm * Fm * 9, w_t2[t], Fm, Fm);
    packW(tw3 + (size_t)t * F * Fm * 9,  w_t3[t], F, Fm);
  }
  packW(post_w, w_post, F, F);
  packW(m1_w, w_m1, 256, 256);
  packW(m2_w, w_m2, 256, 256);
  packW(u1a_w, w_u1a, 128, 256);
  packW(u1b_w, w_u1b, 128, 128);
  packW(u2a_w, w_u2a, F, 128);
  packW(u2b_w, w_u2b, F, F);
  packW(mf_w, w_mf, T, F);
  packW(up1_w, w_up1, F, F);
  packW(up2_w, w_up2, F, F);
  packW(hr_w, w_hr, F, F);
  packW(fin_w, w_fin, 3, F);

  // ---- input layout conversion ----
  {
    int total = B * HW0 * 32;
    nchw2hwc_kernel<<<(total + 255) / 256, 256, 0, stream>>>(in_x, (__bf16*)a_xin, B, 3, 32, HW0);
    total = B * HWs * 256;
    nchw2hwc_kernel<<<(total + 255) / 256, 256, 0, stream>>>(in_spine, (__bf16*)a_sp, B, 256, 256, HWs);
  }

  // ---- multiplexer: attention logits ----
  conv(a_sp, w_m1, m1_b, a_mA, nullptr, Hs, Hs, 256, 256, 0);
  gnSilu(a_mA, m1_g, m1_be, 256, 256, HWs);
  conv(a_mA, w_m2, nullptr, a_mB, nullptr, Hs, Hs, 256, 256, 0);
  gnSilu(a_mB, m2_g, m2_be, 256, 256, HWs);
  up2x(a_mB, a_mu1, 256, Hs, Hs);
  conv(a_mu1, w_u1a, nullptr, a_m3, nullptr, Hm, Hm, 256, 128, 0);
  conv(a_m3, w_u1b, nullptr, a_m4, nullptr, Hm, Hm, 128, 128, 0);
  gnSilu(a_m4, u1b_g, u1b_be, 128, 128, HWm);
  up2x(a_m4, a_mu2, 128, Hm, Hm);
  conv(a_mu2, w_u2a, nullptr, a_m5, nullptr, H0, W0, 128, F, 0);
  conv(a_m5, w_u2b, nullptr, a_m6, nullptr, H0, W0, 64, F, 0);
  gnSilu(a_m6, u2b_g, u2b_be, 64, 64, HW0);
  conv(a_m6, w_mf, nullptr, a_mm, nullptr, H0, W0, 64, T, 0);
  softmax8_kernel<<<(B * HW0 + 255) / 256, 256, 0, stream>>>((const __bf16*)a_mm, (float*)a_att, B * HW0, 32);

  // ---- generator stem ----
  conv(a_xin, w_init, init_b, a_x0, nullptr, H0, W0, 32, F, 0);
  {
    int total = B * HW0 * 64;
    add_noise_kernel<<<(total + 255) / 256, 256, 0, stream>>>((const __bf16*)a_x0, in_noise, noise_sc,
                                                              (__bf16*)a_xs, B, F, 64, HW0);
  }
  conv(a_xs, w_pre, nullptr, a_xs2, nullptr, H0, W0, 64, F, 0);

  // ---- T transform branches + soft switch (incremental combine) ----
  for (int t = 0; t < T; ++t) {
    conv(a_xs2, w_t0[t], nullptr, a_h1, nullptr, H0, W0, 64, Fm, 1);
    conv(a_h1, w_t1[t], nullptr, a_h2, nullptr, H0, W0, 96, Fm, 1);
    conv(a_h2, w_t2[t], nullptr, a_h1, nullptr, H0, W0, 96, Fm, 1);
    conv(a_h1, w_t3[t], nullptr, a_h4, nullptr, H0, W0, 96, F, 0);
    int total = B * HW0 * 64;
    combine_kernel<<<(total + 255) / 256, 256, 0, stream>>>((const __bf16*)a_h4, (const float*)a_att,
                                                            tscale, tbias, t, (float*)a_acc,
                                                            B * HW0, 64, 64, t == 0 ? 1 : 0);
  }
  {
    int total = B * HW0 * 64;
    resid_switch_kernel<<<(total + 255) / 256, 256, 0, stream>>>((const __bf16*)a_x0, (const float*)a_acc,
                                                                 switch_sc, (__bf16*)a_out0, B * HW0, 64, 64);
  }
  conv(a_out0, w_post, post_b, a_pT, nullptr, H0, W0, 64, F, 1);
  {
    int total = B * HW0 * 64;
    resid_scale_kernel<<<(total + 255) / 256, 256, 0, stream>>>((const __bf16*)a_out0, (const __bf16*)a_pT,
                                                                psc_sc, (__bf16*)a_out1, total);
  }

  // ---- upsampling tail ----
  up2x(a_out1, a_u1, 64, H0, W0);
  conv(a_u1, w_up1, up1_b, a_c1, nullptr, H1, H1, 64, F, 1);
  up2x(a_c1, a_u2, 64, H1, H1);
  conv(a_u2, w_up2, up2_b, a_c2, nullptr, H2, H2, 64, F, 1);
  conv(a_c2, w_hr, hr_b, a_c3, nullptr, H2, H2, 64, F, 1);
  conv(a_c3, w_fin, fin_b, nullptr, (float*)d_out, H2, H2, 64, 3, 0);
}
